// Attention_57552561767072
// MI455X (gfx1250) — matrix operand order repaired and checked by reading, not compiled, not run on hardware
//
#include <hip/hip_runtime.h>

// ---------------------------------------------------------------------------
// Attention block for MI455X (gfx1250, wave32, WMMA, async-LDS pipeline).
//   B=2, N=2048, DIM=1024, H=16, HD=64.
// bf16 data path, f32 accumulation via v_wmma_f32_16x16x32_bf16.
// Double-buffered GLOBAL_LOAD_ASYNC_TO_LDS_B128 feeds the WMMA mainloops.
// ---------------------------------------------------------------------------

typedef __attribute__((ext_vector_type(16))) __bf16 v16bf;
typedef __attribute__((ext_vector_type(8)))  float  v8f;

constexpr int kB   = 2;
constexpr int kN   = 2048;
constexpr int kDim = 1024;
constexpr int kH   = 16;
constexpr int kHd  = 64;
constexpr float kEps   = 1e-5f;
constexpr float kScale = 1.0f / (float)kHd;   // reference uses 1/head_dim

// ---------------- WMMA wrapper (8-arg probe-confirmed signature) -----------
__device__ __forceinline__ v8f wmma_bf16(v16bf a, v16bf b, v8f c) {
  return __builtin_amdgcn_wmma_f32_16x16x32_bf16(
      /*neg_a=*/false, a, /*neg_b=*/false, b,
      /*c_mod=*/(short)0, c, /*reuse_a=*/false, /*reuse_b=*/false);
}

// ---------------- async global->LDS copy (CDNA5 ASYNCcnt path) -------------
// Builtin signature (from compiler diagnostic): v4i AS1* src, v4i AS3* lds,
// imm offset, imm cpol.
typedef int v4i __attribute__((vector_size(16)));
typedef __attribute__((address_space(1))) v4i* gv4i_p;
typedef __attribute__((address_space(3))) v4i* lv4i_p;

__device__ __forceinline__ void async_b128(void* lds, const void* g) {
#if __has_builtin(__builtin_amdgcn_global_load_async_to_lds_b128)
  __builtin_amdgcn_global_load_async_to_lds_b128(
      (gv4i_p)(unsigned long long)g, (lv4i_p)lds, 0, 0);
#else
  *reinterpret_cast<uint4*>(lds) = *reinterpret_cast<const uint4*>(g);
#endif
}

__device__ __forceinline__ void wait_async_zero() {
#if __has_builtin(__builtin_amdgcn_s_wait_asynccnt)
  __builtin_amdgcn_s_wait_asynccnt(0);
#else
  asm volatile("s_wait_asynccnt 0" ::: "memory");
#endif
}

// ---------------- fragment loaders (contiguous k order: k = 16*half + elem, wave32) ------------
// A: 16x32 bf16 (MxK), row-major source, leading dim ld (elements).
__device__ __forceinline__ v16bf load_a_frag(const __bf16* a, int ld, int lane) {
  v16bf f;
  const int half = lane >> 4, m = lane & 15;
#pragma unroll
  for (int s = 0; s < 8; ++s) {
    const int kb = (half << 4) + (s << 1);
    f[2 * s]     = a[m * ld + kb];
    f[2 * s + 1] = a[m * ld + kb + 1];
  }
  return f;
}

// B: logical 32x16 (KxN) where B[k][n] = src[n*ld + k]
__device__ __forceinline__ v16bf load_b_fragT(const __bf16* w, int ld, int lane) {
  v16bf f;
  const int half = lane >> 4, n = lane & 15;
#pragma unroll
  for (int s = 0; s < 8; ++s) {
    const int k = (half << 4) + (s << 1);
    f[2 * s]     = w[n * ld + k];
    f[2 * s + 1] = w[n * ld + k + 1];
  }
  return f;
}

// B: logical 32x16 (KxN) where B[k][n] = src[k*ld + n]
__device__ __forceinline__ v16bf load_b_frag(const __bf16* s, int ld, int lane) {
  v16bf f;
  const int half = lane >> 4, n = lane & 15;
#pragma unroll
  for (int t = 0; t < 8; ++t) {
    const int k = (half << 4) + (t << 1);
    f[2 * t]     = s[k * ld + n];
    f[2 * t + 1] = s[(k + 1) * ld + n];
  }
  return f;
}

// ---------------- fp32 -> bf16 conversion (vectorized) ---------------------
__global__ void cvt_f32_bf16(const float* __restrict__ src,
                             __bf16* __restrict__ dst, int n) {
  const int i = (blockIdx.x * blockDim.x + threadIdx.x) * 4;
  if (i < n) {
    const float4 f = *reinterpret_cast<const float4*>(src + i);
    __bf16 o[4] = {(__bf16)f.x, (__bf16)f.y, (__bf16)f.z, (__bf16)f.w};
    *reinterpret_cast<uint2*>(dst + i) = *reinterpret_cast<const uint2*>(o);
  }
}

// ---------------- fused QKV projection: C = x @ W^T ------------------------
// 128x64 block tile, 4 waves; each wave owns two 16-row strips so every
// B fragment feeds two WMMAs. Async double-buffered K loop.
__global__ __launch_bounds__(128) void qkv_gemm_kernel(
    const __bf16* __restrict__ xb,
    const __bf16* __restrict__ wq, const __bf16* __restrict__ wk,
    const __bf16* __restrict__ wv,
    __bf16* __restrict__ q, __bf16* __restrict__ k, __bf16* __restrict__ v) {
  __shared__ __bf16 sA[2][128 * 32];
  __shared__ __bf16 sB[2][64 * 32];
  const int tid = threadIdx.x, lane = tid & 31, wave = tid >> 5;
  const int half = lane >> 4, n16 = lane & 15;
  const int row0 = blockIdx.x * 128, col0 = blockIdx.y * 64;
  const __bf16* w = (blockIdx.z == 0) ? wq : (blockIdx.z == 1) ? wk : wv;
  __bf16* dst     = (blockIdx.z == 0) ? q  : (blockIdx.z == 1) ? k  : v;

  v8f zero = {};
  v8f acc[2][4] = {{zero, zero, zero, zero}, {zero, zero, zero, zero}};

  // A tile: 128 rows x 32 k -> thread t copies row t (4x b128).
  // B tile:  64 rows x 32 k -> thread t copies 16 elems  (2x b128).
  const int brow = tid >> 1, boff = (tid & 1) << 4;
  auto issue = [&](int kt, int buf) {
    const int k0 = kt * 32;
    const __bf16* ga = &xb[(size_t)(row0 + tid) * kDim + k0];
#pragma unroll
    for (int u = 0; u < 4; ++u)
      async_b128(&sA[buf][tid * 32 + u * 8], ga + u * 8);
    const __bf16* gb = &w[(size_t)(col0 + brow) * kDim + k0 + boff];
    async_b128(&sB[buf][brow * 32 + boff],     gb);
    async_b128(&sB[buf][brow * 32 + boff + 8], gb + 8);
  };

  int buf = 0;
  issue(0, 0);
  wait_async_zero();
  __syncthreads();

  for (int kt = 0; kt < kDim / 32; ++kt) {
    if (kt + 1 < kDim / 32) issue(kt + 1, buf ^ 1);
    if (kt + 2 < kDim / 32) {   // L2 prefetch two tiles ahead
      __builtin_prefetch(&xb[(size_t)(row0 + tid) * kDim + kt * 32 + 64], 0, 1);
      __builtin_prefetch(&w[(size_t)(col0 + brow) * kDim + kt * 32 + 64], 0, 1);
    }
    const v16bf af0 = load_a_frag(&sA[buf][(wave * 32) * 32],      32, lane);
    const v16bf af1 = load_a_frag(&sA[buf][(wave * 32 + 16) * 32], 32, lane);
#pragma unroll
    for (int t = 0; t < 4; ++t) {
      const v16bf bf = load_b_fragT(&sB[buf][(t * 16) * 32], 32, lane);
      acc[0][t] = wmma_bf16(af0, bf, acc[0][t]);
      acc[1][t] = wmma_bf16(af1, bf, acc[1][t]);
    }
    wait_async_zero();
    __syncthreads();
    buf ^= 1;
  }

  // scatter C layout into [B,H,N,HD] bf16
#pragma unroll
  for (int s = 0; s < 2; ++s)
#pragma unroll
    for (int t = 0; t < 4; ++t)
#pragma unroll
      for (int r = 0; r < 8; ++r) {
        const int gi = row0 + wave * 32 + s * 16 + half * 8 + r;
        const int gj = col0 + t * 16 + n16;
        const int bb = gi >> 11, nn = gi & (kN - 1);
        const int hh = gj >> 6, dd = gj & 63;
        dst[(((size_t)bb * kH + hh) * kN + nn) * kHd + dd] = (__bf16)acc[s][t][r];
      }
}

// ---------------- RMSNorm + interleaved RoPE, in place on q and k ----------
__global__ __launch_bounds__(256) void rmsnorm_rope_kernel(
    __bf16* __restrict__ q, __bf16* __restrict__ k) {
  const int lane = threadIdx.x & 31, wave = threadIdx.x >> 5;
  const int row  = blockIdx.x * 8 + wave;
  const int BHN  = kB * kH * kN;
  __bf16* base = (row < BHN) ? q : k;
  const int r2 = (row < BHN) ? row : row - BHN;
  __bf16* p = base + (size_t)r2 * kHd;
  const int nn = r2 & (kN - 1);

  float x0 = (float)p[2 * lane], x1 = (float)p[2 * lane + 1];
  float ss = x0 * x0 + x1 * x1;
#pragma unroll
  for (int off = 16; off > 0; off >>= 1) ss += __shfl_xor(ss, off, 32);
  const float inv = rsqrtf(ss * (1.0f / kHd) + kEps);
  x0 *= inv; x1 *= inv;

  const float ang = (float)nn * __powf(10000.0f, -(float)(2 * lane) * (1.0f / kHd));
  float sn, cs; __sincosf(ang, &sn, &cs);
  p[2 * lane]     = (__bf16)(x0 * cs - x1 * sn);
  p[2 * lane + 1] = (__bf16)(x1 * cs + x0 * sn);
}

// ---------------- flash attention per (b,h): 64 q-rows per block -----------
__global__ __launch_bounds__(128) void flash_attn_kernel(
    const __bf16* __restrict__ q, const __bf16* __restrict__ k,
    const __bf16* __restrict__ v, const float* __restrict__ mask,
    __bf16* __restrict__ ob) {
  __shared__ __bf16 sK[2][64 * 64];
  __shared__ __bf16 sV[2][64 * 64];
  __shared__ __bf16 sP[4][16 * 64];   // per-wave P staging (C->A transpose)
  const int tid = threadIdx.x, lane = tid & 31, wave = tid >> 5;
  const int half = lane >> 4, n16 = lane & 15;
  const int bh = blockIdx.y, bb = bh >> 4, hh = bh & (kH - 1);
  const size_t hoff = (size_t)bh * kN * kHd;
  const __bf16* qh = q + hoff;
  const __bf16* kh = k + hoff;
  const __bf16* vh = v + hoff;
  const int q0 = blockIdx.x * 64 + wave * 16;

  // Q strip stays in registers (two K-steps of 32)
  const v16bf aq0 = load_a_frag(qh + (size_t)q0 * kHd,      kHd, lane);
  const v16bf aq1 = load_a_frag(qh + (size_t)q0 * kHd + 32, kHd, lane);

  v8f zero = {};
  v8f accO[4] = {zero, zero, zero, zero};
  float m_r[8], l_r[8];
#pragma unroll
  for (int r = 0; r < 8; ++r) { m_r[r] = -1e30f; l_r[r] = 0.f; }

  const int lrow = tid >> 1, loff = (tid & 1) << 5;   // 64B halves of kv rows
  auto issue_kv = [&](int kb, int buf) {
    const int kv0 = kb * 64;
    const __bf16* gk = &kh[(size_t)(kv0 + lrow) * kHd + loff];
    const __bf16* gv = &vh[(size_t)(kv0 + lrow) * kHd + loff];
#pragma unroll
    for (int u = 0; u < 4; ++u) {
      async_b128(&sK[buf][lrow * 64 + loff + u * 8], gk + u * 8);
      async_b128(&sV[buf][lrow * 64 + loff + u * 8], gv + u * 8);
    }
  };

  int buf = 0;
  issue_kv(0, 0);
  wait_async_zero();
  __syncthreads();

  for (int kb = 0; kb < kN / 64; ++kb) {
    const int kv0 = kb * 64;
    if (kb + 1 < kN / 64) {
      issue_kv(kb + 1, buf ^ 1);
      // next mask block is L2-resident (32MB < 192MB L2); warm it anyway
      __builtin_prefetch(
          &mask[((size_t)bb * kN + q0 + half * 8) * kN + kv0 + 64 + n16], 0, 1);
    }

    // S = Q @ K^T  (wave strip 16x64, f32 accum)
    v8f accS[4];
#pragma unroll
    for (int t = 0; t < 4; ++t) {
      accS[t] = zero;
      accS[t] = wmma_bf16(aq0, load_b_fragT(&sK[buf][(t * 16) * 64],      64, lane), accS[t]);
      accS[t] = wmma_bf16(aq1, load_b_fragT(&sK[buf][(t * 16) * 64 + 32], 64, lane), accS[t]);
    }

    // scale + additive mask; per-row tile max (rows live in 16-lane halves)
    float tmax[8];
#pragma unroll
    for (int r = 0; r < 8; ++r) tmax[r] = -1e30f;
#pragma unroll
    for (int t = 0; t < 4; ++t)
#pragma unroll
      for (int r = 0; r < 8; ++r) {
        const int gi = q0 + half * 8 + r, gj = kv0 + t * 16 + n16;
        const float sv = accS[t][r] * kScale +
                         mask[((size_t)bb * kN + gi) * kN + gj];
        accS[t][r] = sv;
        tmax[r] = fmaxf(tmax[r], sv);
      }
#pragma unroll
    for (int r = 0; r < 8; ++r) {
#pragma unroll
      for (int off = 1; off < 16; off <<= 1)
        tmax[r] = fmaxf(tmax[r], __shfl_xor(tmax[r], off, 32));
    }

    // online softmax update
    float fsc[8], rsum[8];
#pragma unroll
    for (int r = 0; r < 8; ++r) {
      const float mnew = fmaxf(m_r[r], tmax[r]);
      fsc[r] = __expf(m_r[r] - mnew);
      m_r[r] = mnew;
      l_r[r] *= fsc[r];
      rsum[r] = 0.f;
    }
#pragma unroll
    for (int t = 0; t < 4; ++t)
#pragma unroll
      for (int r = 0; r < 8; ++r) {
        accO[t][r] *= fsc[r];
        const float p = __expf(accS[t][r] - m_r[r]);
        rsum[r] += p;
        sP[wave][(half * 8 + r) * 64 + t * 16 + n16] = (__bf16)p;
      }
#pragma unroll
    for (int r = 0; r < 8; ++r) {
#pragma unroll
      for (int off = 1; off < 16; off <<= 1)
        rsum[r] += __shfl_xor(rsum[r], off, 32);
      l_r[r] += rsum[r];
    }

    // O += P @ V   (P re-read from wave-private LDS in A layout)
    const v16bf ap0 = load_a_frag(&sP[wave][0],  64, lane);
    const v16bf ap1 = load_a_frag(&sP[wave][32], 64, lane);
#pragma unroll
    for (int t = 0; t < 4; ++t) {
      accO[t] = wmma_bf16(ap0, load_b_frag(&sV[buf][t * 16],           64, lane), accO[t]);
      accO[t] = wmma_bf16(ap1, load_b_frag(&sV[buf][32 * 64 + t * 16], 64, lane), accO[t]);
    }
    wait_async_zero();
    __syncthreads();
    buf ^= 1;
  }

  // epilogue: 1/l normalize, write merged-head layout [B, N, DIM] bf16
#pragma unroll
  for (int t = 0; t < 4; ++t)
#pragma unroll
    for (int r = 0; r < 8; ++r) {
      const int gi = q0 + half * 8 + r;
      const float invl = (l_r[r] > 0.f) ? 1.0f / l_r[r] : 0.f;
      ob[((size_t)bb * kN + gi) * kDim + hh * kHd + t * 16 + n16] =
          (__bf16)(accO[t][r] * invl);
    }
}

// ---------------- output projection: out = ob @ Wo^T + bo (f32 out) --------
__global__ __launch_bounds__(128) void oproj_kernel(
    const __bf16* __restrict__ ab, const __bf16* __restrict__ wo,
    const float* __restrict__ bo, float* __restrict__ out) {
  __shared__ __bf16 sA[2][128 * 32];
  __shared__ __bf16 sB[2][64 * 32];
  const int tid = threadIdx.x, lane = tid & 31, wave = tid >> 5;
  const int half = lane >> 4, n16 = lane & 15;
  const int row0 = blockIdx.x * 128, col0 = blockIdx.y * 64;

  v8f zero = {};
  v8f acc[2][4] = {{zero, zero, zero, zero}, {zero, zero, zero, zero}};

  const int brow = tid >> 1, boff = (tid & 1) << 4;
  auto issue = [&](int kt, int bufi) {
    const int k0 = kt * 32;
    const __bf16* ga = &ab[(size_t)(row0 + tid) * kDim + k0];
#pragma unroll
    for (int u = 0; u < 4; ++u)
      async_b128(&sA[bufi][tid * 32 + u * 8], ga + u * 8);
    const __bf16* gb = &wo[(size_t)(col0 + brow) * kDim + k0 + boff];
    async_b128(&sB[bufi][brow * 32 + boff],     gb);
    async_b128(&sB[bufi][brow * 32 + boff + 8], gb + 8);
  };

  int buf = 0;
  issue(0, 0);
  wait_async_zero();
  __syncthreads();

  for (int kt = 0; kt < kDim / 32; ++kt) {
    if (kt + 1 < kDim / 32) issue(kt + 1, buf ^ 1);
    if (kt + 2 < kDim / 32) {
      __builtin_prefetch(&ab[(size_t)(row0 + tid) * kDim + kt * 32 + 64], 0, 1);
      __builtin_prefetch(&wo[(size_t)(col0 + brow) * kDim + kt * 32 + 64], 0, 1);
    }
    const v16bf af0 = load_a_frag(&sA[buf][(wave * 32) * 32],      32, lane);
    const v16bf af1 = load_a_frag(&sA[buf][(wave * 32 + 16) * 32], 32, lane);
#pragma unroll
    for (int t = 0; t < 4; ++t) {
      const v16bf bf = load_b_fragT(&sB[buf][(t * 16) * 32], 32, lane);
      acc[0][t] = wmma_bf16(af0, bf, acc[0][t]);
      acc[1][t] = wmma_bf16(af1, bf, acc[1][t]);
    }
    wait_async_zero();
    __syncthreads();
    buf ^= 1;
  }

#pragma unroll
  for (int s = 0; s < 2; ++s)
#pragma unroll
    for (int t = 0; t < 4; ++t)
#pragma unroll
      for (int r = 0; r < 8; ++r) {
        const int gi = row0 + wave * 32 + s * 16 + half * 8 + r;
        const int gj = col0 + t * 16 + n16;
        out[(size_t)gi * kDim + gj] = acc[s][t][r] + bo[gj];
      }
}

// ---------------------------------------------------------------------------
extern "C" void kernel_launch(void* const* d_in, const int* in_sizes, int n_in,
                              void* d_out, int out_size, void* d_ws,
                              size_t ws_size, hipStream_t stream) {
  const float* x    = (const float*)d_in[0];   // [B,N,DIM]
  const float* mask = (const float*)d_in[1];   // [B,N,N]
  const float* Wq   = (const float*)d_in[2];   // [DIM,DIM]
  const float* Wk   = (const float*)d_in[3];
  const float* Wv   = (const float*)d_in[4];
  const float* Wo   = (const float*)d_in[5];
  const float* bo   = (const float*)d_in[6];   // [DIM]
  float* out = (float*)d_out;

  char* ws = (char*)d_ws;
  size_t off = 0;
  auto carve = [&](size_t elems) {
    void* p = ws + off;
    off += (elems * sizeof(__bf16) + 255) & ~(size_t)255;
    return p;
  };
  const size_t nX  = (size_t)kB * kN * kDim;
  const size_t nW  = (size_t)kDim * kDim;
  const size_t nQK = (size_t)kB * kH * kN * kHd;

  __bf16* xb  = (__bf16*)carve(nX);
  __bf16* wqb = (__bf16*)carve(nW);
  __bf16* wkb = (__bf16*)carve(nW);
  __bf16* wvb = (__bf16*)carve(nW);
  __bf16* wob = (__bf16*)carve(nW);
  __bf16* qb  = (__bf16*)carve(nQK);
  __bf16* kbf = (__bf16*)carve(nQK);
  __bf16* vb  = (__bf16*)carve(nQK);
  __bf16* obb = (__bf16*)carve(nX);

  // 1) fp32 -> bf16 conversions (4 elems/thread)
  cvt_f32_bf16<<<(int)((nX / 4 + 255) / 256), 256, 0, stream>>>(x, xb, (int)nX);
  cvt_f32_bf16<<<(int)((nW / 4 + 255) / 256), 256, 0, stream>>>(Wq, wqb, (int)nW);
  cvt_f32_bf16<<<(int)((nW / 4 + 255) / 256), 256, 0, stream>>>(Wk, wkb, (int)nW);
  cvt_f32_bf16<<<(int)((nW / 4 + 255) / 256), 256, 0, stream>>>(Wv, wvb, (int)nW);
  cvt_f32_bf16<<<(int)((nW / 4 + 255) / 256), 256, 0, stream>>>(Wo, wob, (int)nW);

  // 2) fused QKV projections (WMMA), scatter to [B,H,N,HD]
  dim3 g1((kB * kN) / 128, kDim / 64, 3);
  qkv_gemm_kernel<<<g1, 128, 0, stream>>>(xb, wqb, wkb, wvb, qb, kbf, vb);

  // 3) RMSNorm + RoPE in place on q and k
  const int nRows2 = 2 * kB * kH * kN;
  rmsnorm_rope_kernel<<<nRows2 / 8, 256, 0, stream>>>(qb, kbf);

  // 4) flash attention per (b,h)
  dim3 g3(kN / 64, kB * kH);
  flash_attn_kernel<<<g3, 128, 0, stream>>>(qb, kbf, vb, mask, obb);

  // 5) output projection + bias (f32 out)
  dim3 g4((kB * kN) / 128, kDim / 64);
  oproj_kernel<<<g4, 128, 0, stream>>>(obb, wob, bo, out);
}
